// BNN_7988639170688
// MI455X (gfx1250) — 8-bit matrix operand order repaired and checked by reading, compiled, not run on hardware
//
#include <hip/hip_runtime.h>
#include <stdint.h>

typedef int v8i __attribute__((ext_vector_type(8)));

#define B_IMGS 16384
#define A1_BPI 1352
#define OFF_A1  ((size_t)0)
#define SZ_A1   ((size_t)B_IMGS * A1_BPI)
#define OFF_A2  (OFF_A1 + SZ_A1)
#define SZ_A2   ((size_t)B_IMGS * 400)
#define OFF_W1S (((OFF_A2 + SZ_A2) + 255) & ~(size_t)255)
#define SZ_W1S  ((size_t)72 * 4)
#define OFF_B2  (((OFF_W1S + SZ_W1S) + 255) & ~(size_t)255)
#define SZ_B2   ((size_t)16 * 128)
#define OFF_BFC (((OFF_B2 + SZ_B2) + 255) & ~(size_t)255)
#define SZ_BFC  ((size_t)16 * 448)

__device__ __forceinline__ int imax2(int a, int b) { return a > b ? a : b; }

__device__ __forceinline__ int a_k0(int v, int h) {
  return ((v >= 4) ? 32 : 0) + ((v & 2) ? 16 : 0) + ((v & 1) ? 4 : 0) + h * 8;
}
__device__ __forceinline__ int b_k0(int v, int h) {
  return ((v >= 4) ? 32 : 0) + ((v & 2) ? 16 : 0) + ((v & 1) ? 4 : 0) + h * 8;
}

__device__ __forceinline__ int8_t sgnf(float v) {
  return v > 0.f ? (int8_t)1 : (v < 0.f ? (int8_t)-1 : (int8_t)0);
}

__global__ void bnn_prep(const float* __restrict__ w1, const float* __restrict__ w2,
                         const float* __restrict__ wfc,
                         float* __restrict__ w1s, int8_t* __restrict__ b2,
                         int8_t* __restrict__ bfc) {
  int i = blockIdx.x * blockDim.x + threadIdx.x;
  if (i < 72) {
    float w = w1[i];
    w1s[i] = w > 0.f ? 1.f : (w < 0.f ? -1.f : 0.f);
  }
  int j = i - 72;
  if (j >= 0 && j < 16 * 128) {
    int n = j >> 7, k = j & 127;
    int8_t v = 0;
    if (k < 72) {
      int ky = k / 24, rem = k % 24, kx = rem >> 3, ci = rem & 7;
      v = sgnf(w2[((n * 8 + ci) * 3 + ky) * 3 + kx]);
    }
    b2[n * 128 + k] = v;
  }
  int l = i - 72 - 2048;
  if (l >= 0 && l < 16 * 448) {
    int n = l / 448, k = l % 448;
    int8_t v = 0;
    if (n < 10 && k < 400) v = sgnf(wfc[n * 400 + k]);
    bfc[n * 448 + k] = v;
  }
}

__global__ __launch_bounds__(256) void bnn_conv1(const float* __restrict__ x,
                                                 const float* __restrict__ w1s_g,
                                                 int8_t* __restrict__ a1) {
  __shared__ float w1s[72];
  int tid = threadIdx.x;
  if (tid < 72) w1s[tid] = w1s_g[tid];
  __syncthreads();
  int idx = blockIdx.x * 256 + tid;
  int b = idx / 169, o = idx % 169;
  int py = o / 13, px = o % 13;
  const float* xp = x + (size_t)b * 784 + (2 * py) * 28 + 2 * px;
  float win[4][4];
#pragma unroll
  for (int r = 0; r < 4; ++r)
#pragma unroll
    for (int c = 0; c < 4; ++c) win[r][c] = xp[r * 28 + c];

  unsigned long long packed = 0ull;
#pragma unroll
  for (int ch = 0; ch < 8; ++ch) {
    const float* w = &w1s[ch * 9];
    float best = -3.4e38f;
#pragma unroll
    for (int dy = 0; dy < 2; ++dy)
#pragma unroll
      for (int dx = 0; dx < 2; ++dx) {
        float s = 0.f;
#pragma unroll
        for (int ky = 0; ky < 3; ++ky)
#pragma unroll
          for (int kx = 0; kx < 3; ++kx)
            s = fmaf(w[ky * 3 + kx], win[dy + ky][dx + kx], s);
        best = fmaxf(best, s);
      }
    packed |= (unsigned long long)(uint8_t)sgnf(best) << (8 * ch);
  }
  *(unsigned long long*)(a1 + (size_t)idx * 8) = packed;
}

__global__ __launch_bounds__(128) void bnn_conv2(const int8_t* __restrict__ a1,
                                                 const int8_t* __restrict__ b2,
                                                 int8_t* __restrict__ a2) {
  __shared__ int a1s_i[338];
  __shared__ int convs[16 * 128];
  int tid = threadIdx.x;
  int bimg = blockIdx.x;
  const int* src = (const int*)(a1 + (size_t)bimg * A1_BPI);
  for (int i = tid; i < 338; i += 128) a1s_i[i] = src[i];
  __syncthreads();

  const unsigned char* a1b = (const unsigned char*)a1s_i;
  int lane = tid & 31;
  int wave = tid >> 5;
  int h = lane >> 4;
  int mn = lane & 15;

  v8i blo, bhi;
#pragma unroll
  for (int v = 0; v < 8; ++v) {
    int k = b_k0(v, h);
    blo[v] = *(const int*)(b2 + mn * 128 + k);
    bhi[v] = *(const int*)(b2 + mn * 128 + 64 + k);
  }

#pragma unroll
  for (int t = 0; t < 2; ++t) {
    int tile = wave + t * 4;
    int p = tile * 16 + mn;
    bool valid = (p < 121);
    int pc = valid ? p : 0;
    int y = pc / 11;
    int xx = pc % 11;

    v8i alo, ahi;
#pragma unroll
    for (int v = 0; v < 8; ++v) {
      int k = a_k0(v, h);
      {
        int ky = k / 24, rem = k % 24, kx = rem >> 3, c0 = rem & 7;
        int addr = ((y + ky) * 13 + (xx + kx)) * 8 + c0;
        int ld = *(const int*)(a1b + (valid ? addr : 0));
        alo[v] = valid ? ld : 0;
      }
      {
        int kg = 64 + k;
        bool ok = valid && (kg < 72);
        int addr = ((y + 2) * 13 + (xx + 2)) * 8 + (kg - 64);
        int ld = *(const int*)(a1b + (ok ? addr : 0));
        ahi[v] = ok ? ld : 0;
      }
    }
    v8i acc = {};
    acc = __builtin_amdgcn_wmma_i32_16x16x64_iu8(true, alo, true, blo, acc, false, false);
    acc = __builtin_amdgcn_wmma_i32_16x16x64_iu8(true, ahi, true, bhi, acc, false, false);

    int pbase = tile * 16 + 8 * h;
#pragma unroll
    for (int r = 0; r < 8; ++r) convs[mn * 128 + pbase + r] = acc[r];
  }
  __syncthreads();

  for (int o = tid; o < 400; o += 128) {
    int c = o / 25, r2 = o % 25, pyy = r2 / 5, pxx = r2 % 5;
    const int* cv = &convs[c * 128];
    int m0 = cv[(2 * pyy) * 11 + 2 * pxx];
    int m1 = cv[(2 * pyy) * 11 + 2 * pxx + 1];
    int m2 = cv[(2 * pyy + 1) * 11 + 2 * pxx];
    int m3 = cv[(2 * pyy + 1) * 11 + 2 * pxx + 1];
    int mx = imax2(imax2(m0, m1), imax2(m2, m3));
    a2[(size_t)bimg * 400 + o] = mx > 0 ? (int8_t)1 : (mx < 0 ? (int8_t)-1 : (int8_t)0);
  }
}

__global__ __launch_bounds__(256) void bnn_fc(const int8_t* __restrict__ a2,
                                              const int8_t* __restrict__ bfc,
                                              float* __restrict__ out) {
  int tid = threadIdx.x;
  int lane = tid & 31;
  int wave_g = blockIdx.x * 8 + (tid >> 5);
  int row_base = wave_g * 16;
  int h = lane >> 4;
  int mn = lane & 15;
  const int8_t* arow = a2 + (size_t)(row_base + mn) * 400;
  const int8_t* bcol = bfc + mn * 448;

  v8i acc = {};
#pragma unroll
  for (int ch = 0; ch < 7; ++ch) {
    v8i af, bf;
#pragma unroll
    for (int v = 0; v < 8; ++v) {
      int ka = ch * 64 + a_k0(v, h);
      bool ok = (ka < 400);
      int ld = *(const int*)(arow + (ok ? ka : 0));
      af[v] = ok ? ld : 0;
      int kb = ch * 64 + b_k0(v, h);
      bf[v] = *(const int*)(bcol + kb);
    }
    acc = __builtin_amdgcn_wmma_i32_16x16x64_iu8(true, af, true, bf, acc, false, false);
  }
#pragma unroll
  for (int r = 0; r < 8; ++r) {
    int grow = row_base + r + 8 * h;
    if (mn < 10) out[(size_t)grow * 10 + mn] = (float)acc[r];
  }
}

extern "C" void kernel_launch(void* const* d_in, const int* in_sizes, int n_in,
                              void* d_out, int out_size, void* d_ws, size_t ws_size,
                              hipStream_t stream) {
  const float* x   = (const float*)d_in[0];
  const float* w1  = (const float*)d_in[1];
  const float* w2  = (const float*)d_in[2];
  const float* wfc = (const float*)d_in[3];
  char* ws = (char*)d_ws;
  int8_t* A1  = (int8_t*)(ws + OFF_A1);
  int8_t* A2  = (int8_t*)(ws + OFF_A2);
  float*  W1S = (float*)(ws + OFF_W1S);
  int8_t* B2  = (int8_t*)(ws + OFF_B2);
  int8_t* BFC = (int8_t*)(ws + OFF_BFC);
  float* out = (float*)d_out;

  bnn_prep<<<37, 256, 0, stream>>>(w1, w2, wfc, W1S, B2, BFC);
  bnn_conv1<<<(B_IMGS * 169) / 256, 256, 0, stream>>>(x, W1S, A1);
  bnn_conv2<<<B_IMGS, 128, 0, stream>>>(A1, B2, A2);
  bnn_fc<<<128, 256, 0, stream>>>(A2, BFC, out);
}
